// EquivariantNeuralField_67585605370239
// MI455X (gfx1250) — matrix operand order repaired and checked by reading, not compiled, not run on hardware
//
#include <hip/hip_runtime.h>
#include <cstddef>
#include <cstdint>

#define PI_F 3.14159265358979f

// Problem constants
#define BB    2
#define NN    16384
#define LL    256
#define KK    4
#define DINc  2
#define DOUTc 3
#define DLATc 64
#define Hc    128
#define Ac    32
#define NHc   4
#define K1PAD 160   // (H+DIN)=130 padded to multiple of 32

typedef __attribute__((ext_vector_type(16))) _Float16 v16h;
typedef __attribute__((ext_vector_type(8)))  _Float16 v8h;
typedef __attribute__((ext_vector_type(8)))  float    v8f;

static __device__ __forceinline__ void wave_lds_fence() {
  // Wave-private LDS buffers: producer and consumers are lanes of the same
  // wave, so a DScnt drain + compiler barrier is sufficient (no s_barrier).
  asm volatile("s_wait_dscnt 0x0" ::: "memory");
}

// Branch-free gelu: x * sigmoid(1.702-ish tanh-form argument), native v_exp_f32.
static __device__ __forceinline__ float gelu_fast(float x) {
  float u = x * (1.0f + 0.044715f * x * x);   // x + 0.044715 x^3
  return x / (1.0f + __expf(-1.5957691216057308f * u));
}

// ---- WMMA fragment helpers (wave32, V_WMMA_F32_16X16X32_F16 layouts) ----
// A (16xK, f16) from row-major buffer with 'stride' halves per row.
// lanes 0-15: K = kbase+[0..7] then kbase+[16..23]; lanes 16-31: +8.
static __device__ __forceinline__ v16h load_a_frag(const _Float16* base, int stride, int lane) {
  int m  = lane & 15;
  int hi = (lane >> 4) & 1;
  const _Float16* p0 = base + m * stride + (hi ? 8 : 0);
  v8h lo = *(const v8h*)(p0);
  v8h hh = *(const v8h*)(p0 + 16);
  v16h a;
#pragma unroll
  for (int i = 0; i < 8; ++i) { a[i] = lo[i]; a[i + 8] = hh[i]; }
  return a;
}

// B (Kx16 tile, f16) from pre-transposed weights Wt[n][k] row-major (stride halves).
// lanes 0-15: K = kbase+[0..7] then kbase+[16..23]; lanes 16-31: +8.
static __device__ __forceinline__ v16h load_b_frag(const _Float16* base, int stride, int lane) {
  int n  = lane & 15;
  int hi = (lane >> 4) & 1;
  const _Float16* p0 = base + n * stride + (hi ? 8 : 0);
  v8h lo = *(const v8h*)(p0);
  v8h hh = *(const v8h*)(p0 + 16);
  v16h b;
#pragma unroll
  for (int i = 0; i < 8; ++i) { b[i] = lo[i]; b[i + 8] = hh[i]; }
  return b;
}

static __device__ __forceinline__ v8f bcast8(float v) {
  v8f c;
#pragma unroll
  for (int i = 0; i < 8; ++i) c[i] = v;
  return c;
}

static __device__ __forceinline__ v8f wmma_f16(v16h a, v16h b, v8f c) {
  return __builtin_amdgcn_wmma_f32_16x16x32_f16(false, a, false, b, (short)0, c, false, false);
}

// GEMM over a group of NT n-tiles with NKC k-chunks: A fragment is hoisted per
// k-chunk and 8 independent accumulators avoid the WMMA->WMMA D-RAW NOP hazard.
template <int NT, int NKC>
static __device__ __forceinline__ void gemm_group(const _Float16* Abuf, int strideA,
                                                  const _Float16* Bt, int strideB,
                                                  const float* bias, int ntBase,
                                                  int lane, v8f c[NT]) {
  int n16 = lane & 15;
#pragma unroll
  for (int i = 0; i < NT; ++i) c[i] = bcast8(bias[(ntBase + i) * 16 + n16]);
#pragma unroll
  for (int kc = 0; kc < NKC; ++kc) {
    v16h a = load_a_frag(Abuf + kc * 32, strideA, lane);
#pragma unroll
    for (int i = 0; i < NT; ++i) {
      v16h b = load_b_frag(Bt + (size_t)((ntBase + i) * 16) * strideB + kc * 32,
                           strideB, lane);
      c[i] = wmma_f16(a, b, c[i]);
    }
  }
}

// ---------------- weight transpose/convert: src f32 [K][N] -> dst f16 [N][Kpad]
__global__ void ef_wcvt(const float* __restrict__ src, _Float16* __restrict__ dst,
                        int K, int N, int Kpad) {
  int t = blockIdx.x * blockDim.x + threadIdx.x;
  if (t >= N * Kpad) return;
  int n = t / Kpad;
  int k = t - n * Kpad;
  float v = (k < K) ? src[k * N + n] : 0.0f;
  dst[t] = (_Float16)v;
}

// ---------------- stem: cstem = c @ W_stem + b_stem  (512x64 @ 64x128)
__global__ void ef_stem1(const float* __restrict__ cin, const float* __restrict__ W,
                         const float* __restrict__ b, float* __restrict__ cstem) {
  int t = blockIdx.x * blockDim.x + threadIdx.x;   // 65536
  int row = t >> 7, j = t & 127;
  float acc = b[j];
#pragma unroll 8
  for (int d = 0; d < DLATc; ++d) acc += cin[row * DLATc + d] * W[d * Hc + j];
  cstem[t] = acc;
}

// kmat = cstem @ Wk + bk  (512x128 @ 128x128)
__global__ void ef_stem2(const float* __restrict__ cstem, const float* __restrict__ Wk,
                         const float* __restrict__ bk, float* __restrict__ kmat) {
  int t = blockIdx.x * blockDim.x + threadIdx.x;   // 65536
  int row = t >> 7, j = t & 127;
  float acc = bk[j];
#pragma unroll 8
  for (int d = 0; d < Hc; ++d) acc += cstem[row * Hc + d] * Wk[d * Hc + j];
  kmat[t] = acc;
}

// ---------------- kNN: per point, 4 smallest of 256 squared distances
__global__ __launch_bounds__(256) void ef_knn(const float* __restrict__ x,
                                              const float* __restrict__ p,
                                              const float* __restrict__ g,
                                              int*   __restrict__ idxb,
                                              float* __restrict__ biasb,
                                              float* __restrict__ relb) {
  __shared__ float sp[2 * LL];
  __shared__ float sg[LL];
  int tid = threadIdx.x;
  int pid = blockIdx.x * 256 + tid;       // flat point id over B*N, one batch per block
  int b   = pid >> 14;                    // N = 16384
  sp[tid]       = p[b * 2 * LL + tid];
  sp[tid + 256] = p[b * 2 * LL + 256 + tid];
  sg[tid]       = g[b * LL + tid];
  __syncthreads();

  float x0 = x[pid * 2 + 0], x1 = x[pid * 2 + 1];
  float bd0 = 1e30f, bd1 = 1e30f, bd2 = 1e30f, bd3 = 1e30f;
  int   bi0 = 0, bi1 = 0, bi2 = 0, bi3 = 0;
  for (int j = 0; j < LL; ++j) {
    float dx = x0 - sp[2 * j], dy = x1 - sp[2 * j + 1];
    float d = dx * dx + dy * dy;
    if (d < bd3) {
      if (d < bd0)      { bd3=bd2; bi3=bi2; bd2=bd1; bi2=bi1; bd1=bd0; bi1=bi0; bd0=d; bi0=j; }
      else if (d < bd1) { bd3=bd2; bi3=bi2; bd2=bd1; bi2=bi1; bd1=d;  bi1=j; }
      else if (d < bd2) { bd3=bd2; bi3=bi2; bd2=d;  bi2=j; }
      else              { bd3=d;  bi3=j; }
    }
  }
  float bds[4] = {bd0, bd1, bd2, bd3};
  int   bis[4] = {bi0, bi1, bi2, bi3};
#pragma unroll
  for (int kn = 0; kn < KK; ++kn) {
    int rid = pid * KK + kn;
    int li = bis[kn];
    idxb[rid] = li;
    float gg = sg[li];
    biasb[rid] = bds[kn] / (gg * gg);
    relb[2 * rid + 0] = x0 - sp[2 * li + 0];
    relb[2 * rid + 1] = x1 - sp[2 * li + 1];
  }
}

// ---------------- build sin-embedding tile (16 rows x 160 cols, f16 into LDS)
// __sinf -> v_sin_f32 (hardware range reduction; args can be O(100) here).
static __device__ __forceinline__ void build_sin_tile(_Float16* abuf, const float* wsin,
                                                      float cc0, float cc1, int lane) {
#pragma unroll 4
  for (int j = 0; j < 80; ++j) {          // 32 lanes x 80 = 2560 = 16*160
    int t = lane + 32 * j;
    int row = t / K1PAD;
    int col = t - row * K1PAD;
    float r0 = __shfl(cc0, row, 32);
    float r1 = __shfl(cc1, row, 32);
    float val;
    if (col < 2) {
      val = __sinf(col == 0 ? r0 : r1);
    } else if (col < 130) {
      int e = (col - 2) & 63;
      float s = r0 * wsin[e] + r1 * wsin[64 + e];
      if (col >= 66) s += 1.57079632679490f;
      val = __sinf(s);
    } else {
      val = 0.0f;
    }
    abuf[t] = (_Float16)val;
  }
}

// ---------------- mega kernel: emb MLPs + logits + softmax + v-proj + attention
#define NW 8
__global__ __launch_bounds__(256) void ef_mega(
    const float* __restrict__ relb, const float* __restrict__ biasb,
    const int*   __restrict__ idxb,
    const float* __restrict__ cstem, const float* __restrict__ kmat,
    const float* __restrict__ WqSin, const float* __restrict__ WvSin,
    const _Float16* __restrict__ Wq1t, const float* __restrict__ bq1,
    const _Float16* __restrict__ Wq2t, const float* __restrict__ bq2,
    const _Float16* __restrict__ Wv1t, const float* __restrict__ bv1,
    const _Float16* __restrict__ Wv2t, const float* __restrict__ bv2,
    const _Float16* __restrict__ Wvt,  const float* __restrict__ bvv,
    _Float16* __restrict__ ybuf) {
  __shared__ __align__(16) _Float16 sA [NW][16 * K1PAD]; // sin-emb tile
  __shared__ __align__(16) _Float16 sA2[NW][16 * Hc];    // hidden (f16, A-ready)
  __shared__ __align__(16) float    sG [NW][16 * 256];   // f32 scratch (q / vgb)
  __shared__ float sAtt[NW][64];                         // logits -> att
  __shared__ float sWsQ[128], sWsV[128];

  int tid = threadIdx.x, wid = tid >> 5, lane = tid & 31;
  if (tid < 128)       sWsQ[tid]       = WqSin[tid];
  else if (tid < 256)  sWsV[tid - 128] = WvSin[tid - 128];
  __syncthreads();

  int mt = blockIdx.x * NW + wid;        // one 16-row tile per wave
  int rbase = mt * 16;                   // rows are (point, kn) flattened; K=4 | 16
  _Float16* Ab = sA[wid];
  _Float16* A2 = sA2[wid];
  float*    G  = sG[wid];
  float*    AT = sAtt[wid];

  int n16 = lane & 15, hib = (lane >> 4) & 1;

  float cc0 = 0.0f, cc1 = 0.0f;
  if (lane < 16) {
    int rid = rbase + lane;
    cc0 = PI_F * (relb[2 * rid + 0] + 1.0f);
    cc1 = PI_F * (relb[2 * rid + 1] + 1.0f);
  }

  v8f c[8];

  // ======== q path ========
  build_sin_tile(Ab, sWsQ, cc0, cc1, lane);
  wave_lds_fence();
  // layer1: (16x160)x(160x128) + bq1, gelu -> A2
  gemm_group<8, 5>(Ab, K1PAD, Wq1t, K1PAD, bq1, 0, lane, c);
#pragma unroll
  for (int i = 0; i < 8; ++i)
#pragma unroll
    for (int r = 0; r < 8; ++r)
      A2[(r + 8 * hib) * Hc + i * 16 + n16] = (_Float16)gelu_fast(c[i][r]);
  wave_lds_fence();
  // layer2: (16x128)x(128x128) + bq2 -> q (f32 in G, stride 128)
  gemm_group<8, 4>(A2, Hc, Wq2t, Hc, bq2, 0, lane, c);
#pragma unroll
  for (int i = 0; i < 8; ++i)
#pragma unroll
    for (int r = 0; r < 8; ++r)
      G[(r + 8 * hib) * Hc + i * 16 + n16] = c[i][r];
  wave_lds_fence();
  // logits: per (row, head) dot(q, k_gathered) - zx/g^2   -> AT[row*4+h]
#pragma unroll
  for (int rep = 0; rep < 2; ++rep) {
    int row = lane & 15;
    int h   = (lane >> 4) + 2 * rep;
    int rid = rbase + row;
    int li  = idxb[rid];
    int bb  = rid >> 16;                 // (rid/4)/N
    const float* kv = kmat + ((bb << 8) + li) * Hc + h * Ac;
    const float* qv = G + row * Hc + h * Ac;
    float acc = -biasb[rid];
#pragma unroll 8
    for (int a = 0; a < Ac; ++a) acc += qv[a] * kv[a];
    AT[row * 4 + h] = acc;
  }
  wave_lds_fence();
  // softmax over K=4 neighbors, per (point, head): lanes 0..15
  if (lane < 16) {
    int pt = lane >> 2, h = lane & 3;
    float l0 = AT[(pt * 4 + 0) * 4 + h], l1 = AT[(pt * 4 + 1) * 4 + h];
    float l2 = AT[(pt * 4 + 2) * 4 + h], l3 = AT[(pt * 4 + 3) * 4 + h];
    float mx = fmaxf(fmaxf(l0, l1), fmaxf(l2, l3));
    float e0 = __expf(l0 - mx), e1 = __expf(l1 - mx);
    float e2 = __expf(l2 - mx), e3 = __expf(l3 - mx);
    float inv = 1.0f / (e0 + e1 + e2 + e3);
    AT[(pt * 4 + 0) * 4 + h] = e0 * inv;
    AT[(pt * 4 + 1) * 4 + h] = e1 * inv;
    AT[(pt * 4 + 2) * 4 + h] = e2 * inv;
    AT[(pt * 4 + 3) * 4 + h] = e3 * inv;
  }
  wave_lds_fence();

  // ======== v path ========
  build_sin_tile(Ab, sWsV, cc0, cc1, lane);
  wave_lds_fence();
  // layer1 v -> gelu -> A2
  gemm_group<8, 5>(Ab, K1PAD, Wv1t, K1PAD, bv1, 0, lane, c);
#pragma unroll
  for (int i = 0; i < 8; ++i)
#pragma unroll
    for (int r = 0; r < 8; ++r)
      A2[(r + 8 * hib) * Hc + i * 16 + n16] = (_Float16)gelu_fast(c[i][r]);
  wave_lds_fence();
  // layer2 v: (16x128)x(128x256) + bv2 -> vgb (f32 in G, stride 256)
#pragma unroll
  for (int gI = 0; gI < 2; ++gI) {
    gemm_group<8, 4>(A2, Hc, Wv2t, Hc, bv2, gI * 8, lane, c);
#pragma unroll
    for (int i = 0; i < 8; ++i)
#pragma unroll
      for (int r = 0; r < 8; ++r)
        G[(r + 8 * hib) * 256 + (gI * 8 + i) * 16 + n16] = c[i][r];
  }
  wave_lds_fence();
  // vin = c_k * v_g + v_b  -> A2 (f16)
#pragma unroll 4
  for (int j = 0; j < 64; ++j) {         // 32 lanes x 64 = 2048 = 16*128
    int t = lane + 32 * j;
    int m = t >> 7, col = t & 127;
    int rid = rbase + m;
    int li  = idxb[rid];
    int bb  = rid >> 16;
    float ck = cstem[((bb << 8) + li) * Hc + col];
    A2[t] = (_Float16)(ck * G[m * 256 + col] + G[m * 256 + 128 + col]);
  }
  wave_lds_fence();
  // v-proj: (16x128)x(128x512) + bv, fused attention-combine -> y (f16)
#pragma unroll
  for (int gI = 0; gI < 4; ++gI) {       // group gI == head nh
    gemm_group<8, 4>(A2, Hc, Wvt, Hc, bvv, gI * 8, lane, c);
    float attw[8];
#pragma unroll
    for (int r = 0; r < 8; ++r) attw[r] = AT[(r + 8 * hib) * 4 + gI];
    int pt0 = mt * 4 + (hib ? 2 : 0);
#pragma unroll
    for (int i = 0; i < 8; ++i) {
      float p0 = attw[0]*c[i][0] + attw[1]*c[i][1] + attw[2]*c[i][2] + attw[3]*c[i][3];
      float p1 = attw[4]*c[i][4] + attw[5]*c[i][5] + attw[6]*c[i][6] + attw[7]*c[i][7];
      int col = (gI * 8 + i) * 16 + n16;
      ybuf[(size_t)pt0 * 512 + col]       = (_Float16)p0;
      ybuf[(size_t)(pt0 + 1) * 512 + col] = (_Float16)p1;
    }
  }
}

// ---------------- output MLP stage 1: h1 = gelu(y @ Wo1 + bo1), f16
__global__ __launch_bounds__(256) void ef_out1(const _Float16* __restrict__ ybuf,
                                               const _Float16* __restrict__ Wo1t,
                                               const float* __restrict__ bo1,
                                               _Float16* __restrict__ h1) {
  int tid = threadIdx.x, wid = tid >> 5, lane = tid & 31;
  int mt = blockIdx.x * 8 + wid;                   // 2048 tiles of 16 rows
  int n16 = lane & 15, hib = (lane >> 4) & 1;
  const _Float16* Abase = ybuf + (size_t)mt * 16 * 512;
  // speculative prefetch of the activation rows of M-tile mt+8 (same wave, next block)
  __builtin_prefetch(Abase + (size_t)8 * 16 * 512, 0, 1);
  v8f c[8];
#pragma unroll
  for (int ntg = 0; ntg < 4; ++ntg) {
    gemm_group<8, 32>(Abase, 512, Wo1t, 512, bo1, ntg * 8, lane, c);
#pragma unroll
    for (int i = 0; i < 8; ++i)
#pragma unroll
      for (int r = 0; r < 8; ++r)
        h1[(size_t)(mt * 16 + r + 8 * hib) * 512 + (ntg * 8 + i) * 16 + n16] =
            (_Float16)gelu_fast(c[i][r]);
  }
}

// ---------------- output MLP stage 2: out = h1 @ Wo2 + bo2 (N=3, VALU dots)
__global__ __launch_bounds__(256) void ef_out2(const _Float16* __restrict__ h1,
                                               const float* __restrict__ Wo2,
                                               const float* __restrict__ bo2,
                                               float* __restrict__ out) {
  __shared__ float sW[512 * 3];
  __shared__ float sB[3];
  int tid = threadIdx.x;
  for (int i = tid; i < 512 * 3; i += 256) sW[i] = Wo2[i];
  if (tid < 3) sB[tid] = bo2[tid];
  __syncthreads();
  int row = blockIdx.x * 256 + tid;               // 32768 rows
  float a0 = sB[0], a1 = sB[1], a2 = sB[2];
  const _Float16* hp = h1 + (size_t)row * 512;
#pragma unroll 8
  for (int j = 0; j < 512; ++j) {
    float h = (float)hp[j];
    a0 += h * sW[j * 3 + 0];
    a1 += h * sW[j * 3 + 1];
    a2 += h * sW[j * 3 + 2];
  }
  out[row * 3 + 0] = a0;
  out[row * 3 + 1] = a1;
  out[row * 3 + 2] = a2;
}

extern "C" void kernel_launch(void* const* d_in, const int* in_sizes, int n_in,
                              void* d_out, int out_size, void* d_ws, size_t ws_size,
                              hipStream_t stream) {
  const float* x      = (const float*)d_in[0];
  const float* p      = (const float*)d_in[1];
  const float* cin    = (const float*)d_in[2];
  const float* g      = (const float*)d_in[3];
  const float* W_stem = (const float*)d_in[4];
  const float* b_stem = (const float*)d_in[5];
  const float* Wq_sin = (const float*)d_in[6];
  const float* Wq1    = (const float*)d_in[7];
  const float* bq1    = (const float*)d_in[8];
  const float* Wq2    = (const float*)d_in[9];
  const float* bq2    = (const float*)d_in[10];
  const float* Wv_sin = (const float*)d_in[11];
  const float* Wv1    = (const float*)d_in[12];
  const float* bv1    = (const float*)d_in[13];
  const float* Wv2    = (const float*)d_in[14];
  const float* bv2    = (const float*)d_in[15];
  const float* Wk     = (const float*)d_in[16];
  const float* bk     = (const float*)d_in[17];
  const float* Wv     = (const float*)d_in[18];
  const float* bv     = (const float*)d_in[19];
  const float* Wo1    = (const float*)d_in[20];
  const float* bo1    = (const float*)d_in[21];
  const float* Wo2    = (const float*)d_in[22];
  const float* bo2    = (const float*)d_in[23];
  float* out = (float*)d_out;

  // ---- workspace carve
  size_t off = 0;
  char* base = (char*)d_ws;
  auto carve = [&](size_t bytes) -> void* {
    void* r = base + off;
    off += (bytes + 255) & ~(size_t)255;
    return r;
  };
  const size_t NR = (size_t)BB * NN * KK;        // 131072 rows
  const size_t NP = (size_t)BB * NN;             // 32768 points
  float*     cstem = (float*)carve(BB * LL * Hc * sizeof(float));
  float*     kmat  = (float*)carve(BB * LL * Hc * sizeof(float));
  float*     relb  = (float*)carve(NR * 2 * sizeof(float));
  float*     biasb = (float*)carve(NR * sizeof(float));
  int*       idxb  = (int*)carve(NR * sizeof(int));
  _Float16*  Wq1t  = (_Float16*)carve((size_t)Hc * K1PAD * sizeof(_Float16));
  _Float16*  Wq2t  = (_Float16*)carve((size_t)Hc * Hc * sizeof(_Float16));
  _Float16*  Wv1t  = (_Float16*)carve((size_t)Hc * K1PAD * sizeof(_Float16));
  _Float16*  Wv2t  = (_Float16*)carve((size_t)256 * Hc * sizeof(_Float16));
  _Float16*  Wvt   = (_Float16*)carve((size_t)512 * Hc * sizeof(_Float16));
  _Float16*  Wo1t  = (_Float16*)carve((size_t)512 * 512 * sizeof(_Float16));
  _Float16*  ybuf  = (_Float16*)carve(NP * 512 * sizeof(_Float16));
  _Float16*  h1    = (_Float16*)carve(NP * 512 * sizeof(_Float16));
  (void)ws_size; (void)in_sizes; (void)n_in; (void)out_size;

  // ---- weight transpose/convert to f16 [N][Kpad]
  auto wcvt = [&](const float* s, _Float16* d, int K, int N, int Kpad) {
    int n = N * Kpad;
    ef_wcvt<<<(n + 255) / 256, 256, 0, stream>>>(s, d, K, N, Kpad);
  };
  wcvt(Wq1, Wq1t, 130, 128, K1PAD);
  wcvt(Wq2, Wq2t, 128, 128, 128);
  wcvt(Wv1, Wv1t, 130, 128, K1PAD);
  wcvt(Wv2, Wv2t, 128, 256, 128);
  wcvt(Wv,  Wvt,  128, 512, 128);
  wcvt(Wo1, Wo1t, 512, 512, 512);

  // ---- stem
  ef_stem1<<<256, 256, 0, stream>>>(cin, W_stem, b_stem, cstem);
  ef_stem2<<<256, 256, 0, stream>>>(cstem, Wk, bk, kmat);

  // ---- kNN
  ef_knn<<<128, 256, 0, stream>>>(x, p, g, idxb, biasb, relb);

  // ---- fused embedding MLPs + attention (8192 M-tiles, 8 waves/block)
  ef_mega<<<1024, 256, 0, stream>>>(relb, biasb, idxb, cstem, kmat,
                                    Wq_sin, Wv_sin,
                                    Wq1t, bq1, Wq2t, bq2,
                                    Wv1t, bv1, Wv2t, bv2,
                                    Wvt, bv, ybuf);

  // ---- output MLP
  ef_out1<<<256, 256, 0, stream>>>(ybuf, Wo1t, bo1, h1);
  ef_out2<<<128, 256, 0, stream>>>(h1, Wo2, bo2, out);
}
